// TrueBidirectionalMamba_68247030333808
// MI455X (gfx1250) — matrix operand order repaired and checked by reading, not compiled, not run on hardware
//
#include <hip/hip_runtime.h>
#include <cstddef>
#include <cstdint>

// ---------------- problem constants ----------------
#define D_MODEL 1024
#define D_STATE 16
#define D_CONV  4
#define D_INNER 2048
#define DT_RANK 64
#define BATCH   2
#define SEQ     2048
#define NROWS   (BATCH * SEQ)            // 4096 token rows
#define XPROJ_N (DT_RANK + 2 * D_STATE)  // 96
#define CHUNK   64
#define NCHUNK  (SEQ / CHUNK)            // 32

// ---------------- WMMA vector types (gfx1250, wave32) ----------------
typedef __attribute__((ext_vector_type(16))) __bf16 v16bf;
typedef __attribute__((ext_vector_type(8)))  __bf16 v8bf;
typedef __attribute__((ext_vector_type(4)))  __bf16 v4bf;
typedef __attribute__((ext_vector_type(8)))  float  v8f;

__device__ __forceinline__ float softplus_f(float v) {
  return (v > 20.0f) ? v : log1pf(__expf(v));
}
__device__ __forceinline__ float silu_f(float v) {
  return v / (1.0f + __expf(-v));
}

// A fragment: 16 consecutive K values per lane (two adjacent 8-element strips)
__device__ __forceinline__ v16bf load_a_frag(const __bf16* pa, int kc) {
  v8bf lo = *(const v8bf*)(pa + kc);
  v8bf hi = *(const v8bf*)(pa + kc + 8);
  return __builtin_shufflevector(lo, hi, 0, 1, 2, 3, 4, 5, 6, 7,
                                         8, 9, 10, 11, 12, 13, 14, 15);
}
// B fragment: 16 consecutive K values per lane
__device__ __forceinline__ v16bf load_b_frag(const __bf16* pb, int kc) {
  return *(const v16bf*)(pb + kc);
}
__device__ __forceinline__ v8f wmma_bf16(v16bf a, v16bf b, v8f c) {
  return __builtin_amdgcn_wmma_f32_16x16x32_bf16(
      false, a, false, b, (short)0, c, false, false);
}

// =====================================================================
// fp32 -> bf16 conversion (vectorized x4); n must be a multiple of 4
// =====================================================================
__global__ __launch_bounds__(256)
void cvt_bf16_kernel(const float* __restrict__ in, __bf16* __restrict__ out,
                     int n4) {
  const int i = blockIdx.x * 256 + threadIdx.x;
  if (i >= n4) return;
  const float4 v = *(const float4*)(in + (size_t)i * 4);
  v4bf o;
  o[0] = (__bf16)v.x; o[1] = (__bf16)v.y; o[2] = (__bf16)v.z; o[3] = (__bf16)v.w;
  *(v4bf*)(out + (size_t)i * 4) = o;
}

// =====================================================================
// Generic GEMM:  C[M x N] = A[M x K] * W[N x K]^T  (+ bias) (+ softplus)
// A and W are bf16; C is fp32; optional bf16 side-copy of C (cols < nbmax).
// EPI: 0 = none, 1 = +bias, 2 = +bias then softplus
// One wave -> 16x64 tile (4 WMMAs sharing one A fragment); 8 waves/block
// cover 128 rows. K-loop is software-pipelined (load k+32 before WMMA k).
// Requires: M % 128 == 0, N % 32 == 0, K % 32 == 0 (K >= 64).
// =====================================================================
template <int EPI, bool WRITE_BF>
__global__ __launch_bounds__(256)
void gemm_wmma_kernel(const __bf16* __restrict__ A, int lda,
                      const __bf16* __restrict__ W, int ldb,
                      const float* __restrict__ bias,
                      float* __restrict__ C, int ldc,
                      __bf16* __restrict__ Cb, int ldcb, int nbmax,
                      int M, int N, int K) {
  const int lane = threadIdx.x & 31;
  const int wave = threadIdx.x >> 5;
  const int m0 = (blockIdx.y * 8 + wave) * 16;
  const int n0 = blockIdx.x * 64;
  if (m0 >= M) return;
  const bool hiN = (n0 + 32) < N;   // upper 32 cols of the 64-wide tile valid

  const int half = lane >> 4;     // 0: lanes 0-15, 1: lanes 16-31
  const int l16  = lane & 15;

  const __bf16* pa  = A + (size_t)(m0 + l16) * lda + half * 16;
  const __bf16* pb0 = W + (size_t)(n0 + l16)      * ldb + half * 16;
  const __bf16* pb1 = W + (size_t)(n0 + 16 + l16) * ldb + half * 16;
  const __bf16* pb2 = hiN ? W + (size_t)(n0 + 32 + l16) * ldb + half * 16 : pb0;
  const __bf16* pb3 = hiN ? W + (size_t)(n0 + 48 + l16) * ldb + half * 16 : pb1;

  v8f acc0 = {}, acc1 = {}, acc2 = {}, acc3 = {};

  // ---- software-pipelined K loop ----
  v16bf af = load_a_frag(pa, 0);
  v16bf b0 = load_b_frag(pb0, 0);
  v16bf b1 = load_b_frag(pb1, 0);
  v16bf b2 = load_b_frag(pb2, 0);
  v16bf b3 = load_b_frag(pb3, 0);

  for (int kc = 32; kc < K; kc += 32) {
    if (kc + 96 < K) {
      __builtin_prefetch(pa  + kc + 96, 0, 3);   // global_prefetch_b8
      __builtin_prefetch(pb0 + kc + 96, 0, 3);
      __builtin_prefetch(pb1 + kc + 96, 0, 3);
      __builtin_prefetch(pb2 + kc + 96, 0, 3);
      __builtin_prefetch(pb3 + kc + 96, 0, 3);
    }
    // issue next chunk's loads before consuming the current chunk
    v16bf afn = load_a_frag(pa, kc);
    v16bf b0n = load_b_frag(pb0, kc);
    v16bf b1n = load_b_frag(pb1, kc);
    v16bf b2n = load_b_frag(pb2, kc);
    v16bf b3n = load_b_frag(pb3, kc);

    acc0 = wmma_bf16(af, b0, acc0);
    acc1 = wmma_bf16(af, b1, acc1);
    acc2 = wmma_bf16(af, b2, acc2);
    acc3 = wmma_bf16(af, b3, acc3);

    af = afn; b0 = b0n; b1 = b1n; b2 = b2n; b3 = b3n;
  }
  acc0 = wmma_bf16(af, b0, acc0);
  acc1 = wmma_bf16(af, b1, acc1);
  acc2 = wmma_bf16(af, b2, acc2);
  acc3 = wmma_bf16(af, b3, acc3);

  // ---- epilogue: D layout -> row = m0 + 8*half + r, col group + l16 ----
  const int rbase = m0 + half * 8;
  const int c0i = n0 + l16;
  const int c1i = n0 + 16 + l16;
  const int c2i = n0 + 32 + l16;
  const int c3i = n0 + 48 + l16;
  float bia0 = 0.f, bia1 = 0.f, bia2 = 0.f, bia3 = 0.f;
  if (EPI >= 1) {
    bia0 = bias[c0i]; bia1 = bias[c1i];
    if (hiN) { bia2 = bias[c2i]; bia3 = bias[c3i]; }
  }
#pragma unroll
  for (int r = 0; r < 8; ++r) {
    float v0 = acc0[r], v1 = acc1[r], v2 = acc2[r], v3 = acc3[r];
    if (EPI >= 1) { v0 += bia0; v1 += bia1; v2 += bia2; v3 += bia3; }
    if (EPI == 2) {
      v0 = softplus_f(v0); v1 = softplus_f(v1);
      v2 = softplus_f(v2); v3 = softplus_f(v3);
    }
    const size_t row = (size_t)(rbase + r);
    C[row * ldc + c0i] = v0;
    C[row * ldc + c1i] = v1;
    if (hiN) {
      C[row * ldc + c2i] = v2;
      C[row * ldc + c3i] = v3;
    }
    if (WRITE_BF) {
      if (c0i < nbmax) Cb[row * ldcb + c0i] = (__bf16)v0;
      if (c1i < nbmax) Cb[row * ldcb + c1i] = (__bf16)v1;
      if (hiN && c2i < nbmax) Cb[row * ldcb + c2i] = (__bf16)v2;
      if (hiN && c3i < nbmax) Cb[row * ldcb + c3i] = (__bf16)v3;
    }
  }
}

// =====================================================================
// Split xp (NROWS x 2048) into bf16 u_f = xp[:, :1024] and bf16 u_b =
// seq-flip of xp[:, 1024:2048]  (both consumed only by the next GEMM)
// =====================================================================
__global__ __launch_bounds__(256)
void split_flip_kernel(const float* __restrict__ xp,
                       __bf16* __restrict__ uf, __bf16* __restrict__ ub) {
  const size_t idx = (size_t)blockIdx.x * 256 + threadIdx.x; // < NROWS*1024
  const int c   = (int)(idx & (D_MODEL - 1));
  const int row = (int)(idx >> 10);
  const int b = row >> 11;           // / SEQ
  const int t = row & (SEQ - 1);
  uf[idx] = (__bf16)xp[(size_t)row * (2 * D_MODEL) + c];
  const size_t rrow = (size_t)b * SEQ + (SEQ - 1 - t);
  ub[idx] = (__bf16)xp[rrow * (2 * D_MODEL) + D_MODEL + c];
}

// =====================================================================
// Depthwise causal conv (k=4) over seq + SiLU; writes fp32 (for scan)
// and bf16 (for the xproj GEMM).
// =====================================================================
__global__ __launch_bounds__(256)
void conv_silu_kernel(const float* __restrict__ xz,
                      const float* __restrict__ cw,
                      const float* __restrict__ cb,
                      float* __restrict__ out,
                      __bf16* __restrict__ outbf) {
  const int c   = blockIdx.x * 256 + threadIdx.x; // 0..2047
  const int row = blockIdx.y;                      // 0..4095
  const int t   = row & (SEQ - 1);
  float acc = cb[c];
#pragma unroll
  for (int j = 0; j < D_CONV; ++j) {
    const int tt = t - (D_CONV - 1) + j;
    if (tt >= 0)
      acc += cw[c * D_CONV + j] *
             xz[(size_t)(row - (D_CONV - 1) + j) * (2 * D_INNER) + c];
  }
  const float s = silu_f(acc);
  out[(size_t)row * D_INNER + c] = s;
  outbf[(size_t)row * D_INNER + c] = (__bf16)s;
}

// =====================================================================
// Chunked parallel selective scan (associative recurrence h' = a*h + b):
//  Phase 1: per-chunk local scan -> local final state h, decay product P
//  Phase 2: sequential combine over 32 chunks -> true chunk-entry states
//  Phase 3: per-chunk re-scan from true entry state, emit gated y (bf16)
// State layout in ws: [batch][chunk][state][channel]  (coalesced in c)
// =====================================================================
__global__ __launch_bounds__(256)
void scan_chunk_kernel(const float* __restrict__ dt,
                       const float* __restrict__ xdbl,
                       const float* __restrict__ xc,
                       const float* __restrict__ A_log,
                       float* __restrict__ hOut, float* __restrict__ pOut) {
  const int c = blockIdx.x * 256 + threadIdx.x;
  const int q = blockIdx.y;
  const int b = blockIdx.z;
  const int t0 = q * CHUNK;

  __shared__ float sB[CHUNK][D_STATE];
  for (int i = threadIdx.x; i < CHUNK * D_STATE; i += 256) {
    const int tt = i >> 4, s = i & 15;
    sB[tt][s] = xdbl[((size_t)b * SEQ + t0 + tt) * XPROJ_N + DT_RANK + s];
  }
  __syncthreads();

  float Arow[D_STATE], h[D_STATE], P[D_STATE];
#pragma unroll
  for (int s = 0; s < D_STATE; ++s) {
    Arow[s] = -__expf(A_log[(size_t)c * D_STATE + s]);
    h[s] = 0.0f; P[s] = 1.0f;
  }

  for (int tt = 0; tt < CHUNK; ++tt) {
    const size_t row = (size_t)b * SEQ + t0 + tt;
    const float dtv = dt[row * D_INNER + c];
    const float uv  = xc[row * D_INNER + c];
#pragma unroll
    for (int s = 0; s < D_STATE; ++s) {
      const float a = __expf(dtv * Arow[s]);
      h[s] = a * h[s] + dtv * sB[tt][s] * uv;
      P[s] *= a;
    }
  }
  const size_t base = (((size_t)b * NCHUNK + q) * D_STATE) * D_INNER + c;
#pragma unroll
  for (int s = 0; s < D_STATE; ++s) {
    hOut[base + (size_t)s * D_INNER] = h[s];
    pOut[base + (size_t)s * D_INNER] = P[s];
  }
}

__global__ __launch_bounds__(256)
void scan_state_combine_kernel(const float* __restrict__ hC,
                               const float* __restrict__ pC,
                               float* __restrict__ hInit) {
  const int c = blockIdx.x * 256 + threadIdx.x;
  const int b = blockIdx.y;
  float H[D_STATE];
#pragma unroll
  for (int s = 0; s < D_STATE; ++s) H[s] = 0.0f;
  for (int q = 0; q < NCHUNK; ++q) {
    const size_t base = (((size_t)b * NCHUNK + q) * D_STATE) * D_INNER + c;
#pragma unroll
    for (int s = 0; s < D_STATE; ++s) {
      const size_t o = base + (size_t)s * D_INNER;
      hInit[o] = H[s];                       // state entering chunk q
      H[s] = pC[o] * H[s] + hC[o];           // state after chunk q
    }
  }
}

__global__ __launch_bounds__(256)
void scan_final_kernel(const float* __restrict__ dt,
                       const float* __restrict__ xdbl,
                       const float* __restrict__ xc,
                       const float* __restrict__ xz,
                       const float* __restrict__ A_log,
                       const float* __restrict__ Dp,
                       const float* __restrict__ hInit,
                       __bf16* __restrict__ ysbf) {
  const int c = blockIdx.x * 256 + threadIdx.x;
  const int q = blockIdx.y;
  const int b = blockIdx.z;
  const int t0 = q * CHUNK;

  __shared__ float sB[CHUNK][D_STATE];
  __shared__ float sC[CHUNK][D_STATE];
  for (int i = threadIdx.x; i < CHUNK * D_STATE; i += 256) {
    const int tt = i >> 4, s = i & 15;
    const size_t ro = ((size_t)b * SEQ + t0 + tt) * XPROJ_N + DT_RANK;
    sB[tt][s] = xdbl[ro + s];
    sC[tt][s] = xdbl[ro + D_STATE + s];
  }
  __syncthreads();

  float Arow[D_STATE], h[D_STATE];
  const size_t base = (((size_t)b * NCHUNK + q) * D_STATE) * D_INNER + c;
#pragma unroll
  for (int s = 0; s < D_STATE; ++s) {
    Arow[s] = -__expf(A_log[(size_t)c * D_STATE + s]);
    h[s] = hInit[base + (size_t)s * D_INNER];
  }
  const float Dc = Dp[c];

  for (int tt = 0; tt < CHUNK; ++tt) {
    const size_t row = (size_t)b * SEQ + t0 + tt;
    const float dtv = dt[row * D_INNER + c];
    const float uv  = xc[row * D_INNER + c];
    const float zv  = xz[row * (2 * D_INNER) + D_INNER + c];
    float y = 0.0f;
#pragma unroll
    for (int s = 0; s < D_STATE; ++s) {
      const float a = __expf(dtv * Arow[s]);
      h[s] = a * h[s] + dtv * sB[tt][s] * uv;
      y += h[s] * sC[tt][s];
    }
    y += uv * Dc;
    y *= silu_f(zv);
    ysbf[row * D_INNER + c] = (__bf16)y;
  }
}

// =====================================================================
// comb[b,t,:] = yout_f[b,t,:] + yout_b[b, S-1-t, :]   (bf16, feeds GEMM)
// =====================================================================
__global__ __launch_bounds__(256)
void combine_kernel(const float* __restrict__ yf,
                    const float* __restrict__ yb,
                    __bf16* __restrict__ comb) {
  const size_t idx = (size_t)blockIdx.x * 256 + threadIdx.x; // < NROWS*1024
  const int c   = (int)(idx & (D_MODEL - 1));
  const int row = (int)(idx >> 10);
  const int b = row >> 11;
  const int t = row & (SEQ - 1);
  const size_t rrow = (size_t)b * SEQ + (SEQ - 1 - t);
  comb[idx] = (__bf16)(yf[idx] + yb[rrow * D_MODEL + c]);
}

// =====================================================================
// Host-side orchestration
// =====================================================================
extern "C" void kernel_launch(void* const* d_in, const int* in_sizes, int n_in,
                              void* d_out, int out_size, void* d_ws, size_t ws_size,
                              hipStream_t stream) {
  (void)in_sizes; (void)n_in; (void)out_size; (void)ws_size;
  const float* x       = (const float*)d_in[0];
  const float* w_in    = (const float*)d_in[1];
  const float* b_in    = (const float*)d_in[2];
  const float* w_out   = (const float*)d_in[3];
  const float* b_out   = (const float*)d_in[4];
  const float* m_in_w  = (const float*)d_in[5];
  const float* conv_w  = (const float*)d_in[6];
  const float* conv_b  = (const float*)d_in[7];
  const float* xproj_w = (const float*)d_in[8];
  const float* dt_w    = (const float*)d_in[9];
  const float* dt_b    = (const float*)d_in[10];
  const float* A_log   = (const float*)d_in[11];
  const float* Dp      = (const float*)d_in[12];
  const float* m_out_w = (const float*)d_in[13];

  const size_t NR = NROWS;
  const size_t ST = (size_t)BATCH * NCHUNK * D_STATE * D_INNER; // 2M floats

  // ---- fp32 workspace ----
  float* f = (float*)d_ws;
  float* buf_xp   = f;                           // NR*2048
  float* buf_xz   = buf_xp   + NR * 2 * D_MODEL; // NR*4096
  float* buf_conv = buf_xz   + NR * 2 * D_INNER; // NR*2048
  float* buf_xdbl = buf_conv + NR * D_INNER;     // NR*96
  float* buf_dt   = buf_xdbl + NR * XPROJ_N;     // NR*2048
  float* buf_yout = buf_dt   + NR * D_INNER;     // 2*NR*1024
  float* buf_hC   = buf_yout + 2 * NR * D_MODEL; // ST
  float* buf_pC   = buf_hC   + ST;               // ST
  float* buf_hI   = buf_pC   + ST;               // ST
  float* f_end    = buf_hI   + ST;

  // ---- bf16 workspace ----
  __bf16* g = (__bf16*)f_end;
  __bf16* xbf      = g;                                // NR*1024
  __bf16* u_bf     = xbf      + NR * D_MODEL;          // 2*NR*1024
  __bf16* xconv_bf = u_bf     + 2 * NR * D_MODEL;      // NR*2048
  __bf16* xdbl_bf  = xconv_bf + NR * D_INNER;          // NR*64
  __bf16* ys_bf    = xdbl_bf  + NR * DT_RANK;          // NR*2048
  __bf16* comb_bf  = ys_bf    + NR * D_INNER;          // NR*1024
  __bf16* w_in_bf  = comb_bf  + NR * D_MODEL;          // 2048*1024
  __bf16* m_in_bf  = w_in_bf  + 2 * D_MODEL * D_MODEL; // 4096*1024
  __bf16* xproj_bf = m_in_bf  + 2 * (size_t)D_INNER * D_MODEL;
  __bf16* dt_w_bf  = xproj_bf + (size_t)XPROJ_N * D_INNER;
  __bf16* m_out_bf = dt_w_bf  + (size_t)D_INNER * DT_RANK;
  __bf16* w_out_bf = m_out_bf + (size_t)D_MODEL * D_INNER;

  const dim3 blk(256);
  auto cvt = [&](const float* src, __bf16* dst, size_t n) {
    const int n4 = (int)(n / 4);
    cvt_bf16_kernel<<<dim3((n4 + 255) / 256), blk, 0, stream>>>(src, dst, n4);
  };
  auto gx = [](int N) { return (unsigned)((N + 63) / 64); };

  // 0) one-time bf16 conversions (weights + input activations)
  cvt(x,       xbf,      NR * D_MODEL);
  cvt(w_in,    w_in_bf,  (size_t)2 * D_MODEL * D_MODEL);
  cvt(m_in_w,  m_in_bf,  (size_t)2 * D_INNER * D_MODEL);
  cvt(xproj_w, xproj_bf, (size_t)XPROJ_N * D_INNER);
  cvt(dt_w,    dt_w_bf,  (size_t)D_INNER * DT_RANK);
  cvt(m_out_w, m_out_bf, (size_t)D_MODEL * D_INNER);
  cvt(w_out,   w_out_bf, (size_t)D_MODEL * D_MODEL);

  // 1) xp = x @ w_in^T + b_in                 (4096 x 2048, K=1024)
  gemm_wmma_kernel<1, false><<<dim3(gx(2 * D_MODEL), NROWS / 128), blk, 0, stream>>>(
      xbf, D_MODEL, w_in_bf, D_MODEL, b_in, buf_xp, 2 * D_MODEL,
      nullptr, 0, 0, NROWS, 2 * D_MODEL, D_MODEL);

  // 2) split into bf16 u_f and seq-flipped bf16 u_b
  split_flip_kernel<<<dim3((unsigned)(NR * D_MODEL / 256)), blk, 0, stream>>>(
      buf_xp, u_bf, u_bf + NR * D_MODEL);

  // 3) per-direction Mamba block
  for (int dir = 0; dir < 2; ++dir) {
    const __bf16* u_d = u_bf + (size_t)dir * NR * D_MODEL;

    // xz = u @ m_in_w^T                      (4096 x 4096, K=1024)
    gemm_wmma_kernel<0, false><<<dim3(gx(2 * D_INNER), NROWS / 128), blk, 0, stream>>>(
        u_d, D_MODEL, m_in_bf, D_MODEL, nullptr, buf_xz, 2 * D_INNER,
        nullptr, 0, 0, NROWS, 2 * D_INNER, D_MODEL);

    // xconv = silu(depthwise_conv(xc) + conv_b)  -> fp32 + bf16
    conv_silu_kernel<<<dim3(D_INNER / 256, NROWS), blk, 0, stream>>>(
        buf_xz, conv_w, conv_b, buf_conv, xconv_bf);

    // x_dbl = xconv @ xproj_w^T              (4096 x 96, K=2048)
    // fp32 for scan (B/C), bf16 copy of first 64 cols for dt GEMM
    gemm_wmma_kernel<0, true><<<dim3(gx(XPROJ_N), NROWS / 128), blk, 0, stream>>>(
        xconv_bf, D_INNER, xproj_bf, D_INNER, nullptr, buf_xdbl, XPROJ_N,
        xdbl_bf, DT_RANK, DT_RANK, NROWS, XPROJ_N, D_INNER);

    // dt = softplus(x_dbl[:, :64] @ dt_w^T + dt_b)   (4096 x 2048, K=64)
    gemm_wmma_kernel<2, false><<<dim3(gx(D_INNER), NROWS / 128), blk, 0, stream>>>(
        xdbl_bf, DT_RANK, dt_w_bf, DT_RANK, dt_b, buf_dt, D_INNER,
        nullptr, 0, 0, NROWS, D_INNER, DT_RANK);

    // chunked selective scan (3 phases), fused Dp skip + silu(z) gate
    scan_chunk_kernel<<<dim3(D_INNER / 256, NCHUNK, BATCH), blk, 0, stream>>>(
        buf_dt, buf_xdbl, buf_conv, A_log, buf_hC, buf_pC);
    scan_state_combine_kernel<<<dim3(D_INNER / 256, BATCH), blk, 0, stream>>>(
        buf_hC, buf_pC, buf_hI);
    scan_final_kernel<<<dim3(D_INNER / 256, NCHUNK, BATCH), blk, 0, stream>>>(
        buf_dt, buf_xdbl, buf_conv, buf_xz, A_log, Dp, buf_hI, ys_bf);

    // yout_dir = y @ m_out_w^T               (4096 x 1024, K=2048)
    gemm_wmma_kernel<0, false><<<dim3(gx(D_MODEL), NROWS / 128), blk, 0, stream>>>(
        ys_bf, D_INNER, m_out_bf, D_INNER, nullptr,
        buf_yout + (size_t)dir * NR * D_MODEL, D_MODEL,
        nullptr, 0, 0, NROWS, D_MODEL, D_INNER);
  }

  // 4) comb = out_f + flip(out_b)   (bf16)
  combine_kernel<<<dim3((unsigned)(NR * D_MODEL / 256)), blk, 0, stream>>>(
      buf_yout, buf_yout + NR * D_MODEL, comb_bf);

  // 5) out = comb @ w_out^T + b_out          (4096 x 1024, K=1024)
  gemm_wmma_kernel<1, false><<<dim3(gx(D_MODEL), NROWS / 128), blk, 0, stream>>>(
      comb_bf, D_MODEL, w_out_bf, D_MODEL, b_out, (float*)d_out, D_MODEL,
      nullptr, 0, 0, NROWS, D_MODEL, D_MODEL);
}
